// SingleHeadDilatedSelfAttention_1786706395705
// MI455X (gfx1250) — matrix operand order repaired and checked by reading, not compiled, not run on hardware
//
#include <hip/hip_runtime.h>

namespace {
constexpr int kB   = 4;
constexpr int kT   = 8192;
constexpr int kE   = 1024;
constexpr int kSeg = 16;
constexpr int kL   = 256;   // dilated segments (n_seg / DIL)
constexpr int kDk  = 64;
constexpr float kNorm   = 0.125f;     // 1/sqrt(64)
constexpr float kBigNeg = -10000.0f;
constexpr int kFragHalves = 12 * 32 * 16;   // one k-step of packed W = 12 KB
}

typedef __attribute__((ext_vector_type(16))) _Float16 v16h;
typedef __attribute__((ext_vector_type(8)))  _Float16 v8h;
typedef __attribute__((ext_vector_type(8)))  float    v8f;

static __device__ __forceinline__ v8f zero8() {
  v8f z;
#pragma unroll
  for (int i = 0; i < 8; ++i) z[i] = 0.0f;
  return z;
}

static __device__ __forceinline__ void cvt8(const float* __restrict__ p, _Float16* d) {
  float4 u = *(const float4*)p;
  float4 v = *(const float4*)(p + 4);
  d[0] = (_Float16)u.x; d[1] = (_Float16)u.y; d[2] = (_Float16)u.z; d[3] = (_Float16)u.w;
  d[4] = (_Float16)v.x; d[5] = (_Float16)v.y; d[6] = (_Float16)v.z; d[7] = (_Float16)v.w;
}

// A fragment (16x32 f16, MxK), ISA 7.12.2 layout: lanes<16 take K = {0..7,16..23},
// lanes 16-31 take K = {8..15,24..31}. Source row is fp32.
static __device__ __forceinline__ v16h load_a_from_f32(const float* __restrict__ row,
                                                       int k0, int koff) {
  _Float16 t[16];
  cvt8(row + k0 + koff,      t);
  cvt8(row + k0 + 16 + koff, t + 8);
  v16h a;
#pragma unroll
  for (int i = 0; i < 16; ++i) a[i] = t[i];
  return a;
}

static __device__ __forceinline__ v16h cat8(v8h lo, v8h hi) {
  v16h r;
#pragma unroll
  for (int i = 0; i < 8; ++i) { r[i] = lo[i]; r[i + 8] = hi[i]; }
  return r;
}

// --------------------------------------------------------------------------
// Kernel 0: zero the output (odd segments must be zero; even ones overwritten).
// --------------------------------------------------------------------------
__global__ void zero_f4_kernel(float4* __restrict__ p, int n4) {
  int i = blockIdx.x * blockDim.x + threadIdx.x;
  if (i < n4) p[i] = make_float4(0.f, 0.f, 0.f, 0.f);
}

// --------------------------------------------------------------------------
// Kernel P: pack Wq/Wk/Wv into per-lane B-fragment order (f16), k-step major:
//   Wfrag[(ks*12 + mat*4 + nt)*32 + lane][0..15] = W[nt*16+ln][ks*32 + kb + (i<8 ? i : i+8)]
// so the 12 fragments of one k-step form one contiguous 12 KB span.
//   grid = 32*12 blocks of 32 lanes.
// --------------------------------------------------------------------------
__global__ __launch_bounds__(32) void pack_w_kernel(
    const float* __restrict__ Wq,
    const float* __restrict__ Wk,
    const float* __restrict__ Wv,
    _Float16* __restrict__ Wfrag) {
  const int blk  = blockIdx.x;            // ks*12 + mat*4 + nt
  const int nt   = blk & 3;
  const int mat  = (blk >> 2) % 3;
  const int ks   = blk / 12;
  const int lane = threadIdx.x;
  const int ln   = lane & 15;
  const int kb   = (lane < 16) ? 0 : 8;   // B-frag K base

  const float* __restrict__ W = (mat == 0) ? Wq : (mat == 1) ? Wk : Wv;
  const float* __restrict__ src = W + (size_t)(nt * 16 + ln) * kE + ks * 32 + kb;
  _Float16* __restrict__ dst = Wfrag + ((size_t)blk * 32 + lane) * 16;
#pragma unroll
  for (int i = 0; i < 16; ++i) dst[i] = (_Float16)src[(i < 8) ? i : i + 8];
}

// --------------------------------------------------------------------------
// Kernel 1: QKV projection. Cooperative 4-wave workgroup:
//   - each wave owns one 16-token tile (tile = blockIdx.x*4 + wave)
//   - the 12 KB of packed W fragments per k-step is staged global->LDS by all
//     128 threads into a ping-pong buffer, so stage(ks+1) overlaps WMMA(ks);
//   - B operands are then short-latency ds_load_b128 instead of L2 round-trips.
// --------------------------------------------------------------------------
__global__ __launch_bounds__(128) void qkv_proj_kernel(
    const float* __restrict__ x,
    const _Float16* __restrict__ Wfrag,   // packed B fragments, k-step major
    _Float16* __restrict__ Qb,
    _Float16* __restrict__ Kb,
    _Float16* __restrict__ Vb) {
  __shared__ __attribute__((aligned(32))) _Float16 Wlds[2][kFragHalves];  // 2 x 12 KB

  const int tid  = threadIdx.x;
  const int wid  = tid >> 5;
  const int lane = tid & 31;
  const int ln   = lane & 15;
  const int koff = (lane < 16) ? 0 : 8;   // A-frag K sub-base

  const int tile = blockIdx.x * 4 + wid;  // b*kL + l
  const float* __restrict__ xrow =
      x + ((size_t)(tile >> 8) * kT + 32 * (tile & (kL - 1)) + ln) * kE;

  // cooperative stage of one k-step's fragment span (768 x 16 B)
  auto stage = [&](int ks, int buf) {
    const uint4* __restrict__ src =
        (const uint4*)(Wfrag + (size_t)ks * kFragHalves);
    uint4* __restrict__ dst = (uint4*)&Wlds[buf][0];
    uint4 t[6];
#pragma unroll
    for (int i = 0; i < 6; ++i) t[i] = src[tid + i * 128];
#pragma unroll
    for (int i = 0; i < 6; ++i) dst[tid + i * 128] = t[i];
  };

  v8f acc[12];
#pragma unroll
  for (int q = 0; q < 12; ++q) acc[q] = zero8();

  stage(0, 0);
  v16h a = load_a_from_f32(xrow, 0, koff);
  __syncthreads();

  for (int ks = 0; ks < 32; ++ks) {
    __builtin_prefetch(xrow + ks * 32 + 256, 0, 3);   // global_prefetch_b8, near

    if (ks < 31) stage(ks + 1, (ks + 1) & 1);         // overlaps this step's WMMAs

    v16h an = a;
    if (ks < 31) an = load_a_from_f32(xrow, (ks + 1) * 32, koff);

    const _Float16* __restrict__ wl = Wlds[ks & 1];
#pragma unroll
    for (int q = 0; q < 12; ++q) {
      const v16h bm = *(const v16h*)&wl[(q * 32 + lane) * 16];
      acc[q] = __builtin_amdgcn_wmma_f32_16x16x32_f16(
          false, a, false, bm, (short)0, acc[q], false, false);
    }
    a = an;
    __syncthreads();   // stage(ks+1) visible; acc reads done before next overwrite
  }

  const int mrow = (lane < 16) ? 0 : 8;   // C layout: M = vgpr + 8*(lane>=16)
#pragma unroll
  for (int q = 0; q < 12; ++q) {
    _Float16* __restrict__ out = (q < 4) ? Qb : (q < 8) ? Kb : Vb;
    const int nt = q & 3;
#pragma unroll
    for (int m = 0; m < 8; ++m) {
      out[((size_t)tile * 16 + (m + mrow)) * kDk + nt * 16 + ln] =
          (_Float16)acc[q][m];
    }
  }
}

// --------------------------------------------------------------------------
// Kernel 2: dilated causal attention, one wave per (b, j, 16-query o-tile).
//   S[l,o] = K[l,:]·Q[o,:];  mask l<=o;  softmax over l;  att = P^T @ V.
// --------------------------------------------------------------------------
__global__ __launch_bounds__(32) void dilated_attn_kernel(
    const _Float16* __restrict__ Qb,
    const _Float16* __restrict__ Kb,
    const _Float16* __restrict__ Vb,
    float* __restrict__ out) {
  __shared__ float    Slds[kL][16];                                     // 16 KB scores [l][o]
  __shared__ __attribute__((aligned(32))) _Float16 Plds[16][kL + 32];   //  9 KB probs^T [o][l]
  __shared__ __attribute__((aligned(32))) _Float16 VldsT[kDk][kL + 16]; // 34 KB V^T [v][l]

  const int ot   = blockIdx.x;            // query tile
  const int j    = blockIdx.y;            // intra-segment position
  const int b    = blockIdx.z;
  const int lane = threadIdx.x;
  const int ln   = lane & 15;
  const int koff = (lane < 16) ? 0 : 8;
  const int kb   = (lane < 16) ? 0 : 16;
  const int mrow = (lane < 16) ? 0 : 8;

  const int o0  = ot * 16;
  const int nlb = ot + 1;                 // causal: key blocks with l <= o
  const int nl  = nlb * 16;

  // Q tile as the B operand of S = K @ Q^T (N = query o, K-dim = d).
  const _Float16* __restrict__ qrow = Qb + (((size_t)(b * kL + o0 + ln) * 16 + j) * kDk);
  const v16h bq0 = cat8(*(const v8h*)(qrow + koff),      *(const v8h*)(qrow + 16 + koff));
  const v16h bq1 = cat8(*(const v8h*)(qrow + 32 + koff), *(const v8h*)(qrow + 48 + koff));

  for (int lb = 0; lb < nlb; ++lb) {
    const _Float16* __restrict__ krow =
        Kb + (((size_t)(b * kL + lb * 16 + ln) * 16 + j) * kDk);
    v16h a0 = cat8(*(const v8h*)(krow + koff),      *(const v8h*)(krow + 16 + koff));
    v16h a1 = cat8(*(const v8h*)(krow + 32 + koff), *(const v8h*)(krow + 48 + koff));
    v8f s = zero8();
    s = __builtin_amdgcn_wmma_f32_16x16x32_f16(false, a0, false, bq0, (short)0, s, false, false);
    s = __builtin_amdgcn_wmma_f32_16x16x32_f16(false, a1, false, bq1, (short)0, s, false, false);
#pragma unroll
    for (int m = 0; m < 8; ++m) {
      const int l = lb * 16 + m + mrow;
      const int o = o0 + ln;
      Slds[l][ln] = ((l <= o) ? s[m] : kBigNeg) * kNorm;   // mask then *norm, as reference
    }
    // stage V^T for this key block (B-operand-friendly: contiguous K per lane)
    for (int idx = lane; idx < 16 * kDk; idx += 32) {
      const int r = idx >> 6;
      const int v = idx & 63;
      VldsT[v][lb * 16 + r] =
          Vb[(((size_t)(b * kL + lb * 16 + r) * 16 + j) * kDk) + v];
    }
  }

  if (nlb & 1) {  // pad P@V K-dim to a multiple of 32 with zeros
    for (int idx = lane; idx < 16 * 16; idx += 32)
      Plds[idx >> 4][nl + (idx & 15)] = (_Float16)0.0f;
    for (int idx = lane; idx < kDk * 16; idx += 32)
      VldsT[idx >> 4][nl + (idx & 15)] = (_Float16)0.0f;
  }
  __syncthreads();

  // softmax over l for column o = o0 + ln; lane pair (n, n+16) splits the rows.
  {
    const int start = (lane < 16) ? 0 : 1;
    float mx = -3.0e38f;
    for (int l = start; l < nl; l += 2) mx = fmaxf(mx, Slds[l][ln]);
    mx = fmaxf(mx, __shfl_xor(mx, 16, 32));
    float sum = 0.0f;
    for (int l = start; l < nl; l += 2) sum += __expf(Slds[l][ln] - mx);
    sum += __shfl_xor(sum, 16, 32);
    const float inv = 1.0f / sum;
    for (int l = start; l < nl; l += 2)
      Plds[ln][l] = (_Float16)(__expf(Slds[l][ln] - mx) * inv);
  }
  __syncthreads();

  // att = P^T @ V : A = P^T (16 o x K l), B = V (K l x 16 v), 4 tiles cover v=64.
  v8f acc[4];
#pragma unroll
  for (int vt = 0; vt < 4; ++vt) acc[vt] = zero8();

  const int npair = (nlb + 1) >> 1;
  for (int kp = 0; kp < npair; ++kp) {
    const int lb0 = kp * 32;
    v16h a = cat8(*(const v8h*)&Plds[ln][lb0 + koff],
                  *(const v8h*)&Plds[ln][lb0 + 16 + koff]);
#pragma unroll
    for (int vt = 0; vt < 4; ++vt) {
      v16h bv = cat8(*(const v8h*)&VldsT[vt * 16 + ln][lb0 + koff],
                     *(const v8h*)&VldsT[vt * 16 + ln][lb0 + 16 + koff]);
      acc[vt] = __builtin_amdgcn_wmma_f32_16x16x32_f16(
          false, a, false, bv, (short)0, acc[vt], false, false);
    }
  }

  // scatter: res[b, 32*o + j, v] = att[o, v]   (padding='same' strided write)
#pragma unroll
  for (int vt = 0; vt < 4; ++vt) {
#pragma unroll
    for (int m = 0; m < 8; ++m) {
      const int o = o0 + m + mrow;
      const int t = 32 * o + j;
      out[((size_t)b * kT + t) * kDk + vt * 16 + ln] = acc[vt][m];
    }
  }
}

// --------------------------------------------------------------------------
extern "C" void kernel_launch(void* const* d_in, const int* in_sizes, int n_in,
                              void* d_out, int out_size, void* d_ws, size_t ws_size,
                              hipStream_t stream) {
  (void)in_sizes; (void)n_in; (void)ws_size;
  const float* x  = (const float*)d_in[0];
  const float* Wk = (const float*)d_in[1];
  const float* Wq = (const float*)d_in[2];
  const float* Wv = (const float*)d_in[3];
  float* out = (float*)d_out;

  const size_t rows = (size_t)kB * kL * kSeg;     // 16384 projected tokens
  _Float16* Qb    = (_Float16*)d_ws;              // 2 MB each
  _Float16* Kb    = Qb + rows * kDk;
  _Float16* Vb    = Kb + rows * kDk;
  _Float16* Wfrag = Vb + rows * kDk;              // 384 KB packed W fragments

  const int n4 = out_size / 4;
  zero_f4_kernel<<<(n4 + 255) / 256, 256, 0, stream>>>((float4*)d_out, n4);
  pack_w_kernel<<<32 * 12, 32, 0, stream>>>(Wq, Wk, Wv, Wfrag);
  qkv_proj_kernel<<<kB * kL / 4, 128, 0, stream>>>(x, Wfrag, Qb, Kb, Vb);
  dilated_attn_kernel<<<dim3(kL / 16, kSeg, kB), 32, 0, stream>>>(Qb, Kb, Vb, out);
}
